// DCN_19902878449871
// MI455X (gfx1250) — compile-verified, matrix operand order checked by reading, not run on hardware
//
#include <hip/hip_runtime.h>

typedef __attribute__((ext_vector_type(16))) _Float16 v16h;
typedef __attribute__((ext_vector_type(8)))  float    v8f;

union Frag16 { v16h v; unsigned u[8]; };

#define ROWS 32
#define XS   528   // LDS row stride (f16 elems) for x0 tile (512 + pad)
#define H1S  272   // 256 + pad
#define H2S  144   // 128 + pad

__device__ __forceinline__ float wave_reduce(float v) {
#pragma unroll
  for (int m = 16; m >= 1; m >>= 1) v += __shfl_xor(v, m, 32);
  return v;
}

// Load a 16x32 f16 fragment (A-style layout per CDNA5 ISA 7.12.2) from a
// row-major [rows][stride] f16 buffer. Also used for B with weights stored
// pre-transposed [N][K] so lane = n, pairs along K.
__device__ __forceinline__ v16h load_frag_f16(const _Float16* __restrict__ base,
                                              int row0, int kbase, int stride,
                                              int lane) {
  const int m  = lane & 15;
  const int hi = (lane >> 4) & 1;
  const _Float16* p = base + (long)(row0 + m) * stride + kbase + hi * 8;
  Frag16 f;
#pragma unroll
  for (int j = 0; j < 4; ++j)          // K = kbase + hi*8 + {0,2,4,6}
    f.u[j] = *(const unsigned*)(p + 2 * j);
#pragma unroll
  for (int j = 0; j < 4; ++j)          // K = kbase + 16 + hi*8 + {0,2,4,6}
    f.u[4 + j] = *(const unsigned*)(p + 16 + 2 * j);
  return f.v;
}

__device__ __forceinline__ v8f wmma_f16(v16h a, v16h b, v8f c) {
  return __builtin_amdgcn_wmma_f32_16x16x32_f16(
      /*neg_a=*/false, a, /*neg_b=*/false, b,
      /*c_mod=*/(short)0, c, /*reuse_a=*/false, /*reuse_b=*/false);
}

// fp32 [K][N] -> f16 transposed [N][K]
__global__ void cvt_transpose(const float* __restrict__ in,
                              _Float16* __restrict__ out, int K, int N) {
  int idx = blockIdx.x * blockDim.x + threadIdx.x;
  if (idx >= K * N) return;
  int n = idx / K;
  int k = idx - n * K;
  out[(long)n * K + k] = (_Float16)in[(long)k * N + n];
}

__global__ __launch_bounds__(256)
void dcn_fused(const int* __restrict__ user, const int* __restrict__ item,
               const float* __restrict__ user_emb, const float* __restrict__ item_emb,
               const float* __restrict__ cw1, const float* __restrict__ cw2,
               const _Float16* __restrict__ W1t, const float* __restrict__ db1,
               const _Float16* __restrict__ W2t, const float* __restrict__ db2,
               const _Float16* __restrict__ W3t, const float* __restrict__ db3,
               const float* __restrict__ Wo, const float* __restrict__ bo,
               float* __restrict__ out) {
  __shared__ _Float16 x0h[ROWS * XS];
  __shared__ _Float16 h1[ROWS * H1S];
  __shared__ _Float16 h2[ROWS * H2S];
  __shared__ float rowAcc[ROWS];

  const int tid  = threadIdx.x;
  const int wave = tid >> 5;
  const int lane = tid & 31;
  const int hi   = (lane >> 4) & 1;
  const int l16  = lane & 15;
  const int rowBase = blockIdx.x * ROWS;

  // ---------- Phase A: embedding gather + cross layers + x2.Wo partial ----
  float cwl1[16], cwl2[16], wol[16];
#pragma unroll
  for (int j = 0; j < 16; ++j) {
    cwl1[j] = cw1[lane * 16 + j];
    cwl2[j] = cw2[lane * 16 + j];
    wol[j]  = Wo[lane * 16 + j];
  }

  for (int rr = 0; rr < 4; ++rr) {       // 8 waves x 4 rows = 32 rows
    const int row = wave * 4 + rr;
    const int g   = rowBase + row;
    const int u   = user[g];
    const int it  = item[g];
    const float* src = (lane < 16) ? (user_emb + (long)u  * 256 + l16 * 16)
                                   : (item_emb + (long)it * 256 + l16 * 16);
    float x0[16];
#pragma unroll
    for (int j = 0; j < 16; ++j) x0[j] = src[j];

    float d1 = 0.f;
#pragma unroll
    for (int j = 0; j < 16; ++j) d1 += x0[j] * cwl1[j];
    d1 = wave_reduce(d1);                 // x0 . cw1

    float x1[16];
#pragma unroll
    for (int j = 0; j < 16; ++j) x1[j] = x0[j] * d1 + x0[j];

    float d2 = 0.f;
#pragma unroll
    for (int j = 0; j < 16; ++j) d2 += x1[j] * cwl2[j];
    d2 = wave_reduce(d2);                 // x1 . cw2

    float p = 0.f;
#pragma unroll
    for (int j = 0; j < 16; ++j) {
      float x2 = x0[j] * d2 + x1[j];
      p += x2 * wol[j];                   // x2 . Wo[0:512]
    }
    p = wave_reduce(p);
    if (lane == 0) rowAcc[row] = p;

    _Float16* dst = x0h + row * XS + lane * 16;
#pragma unroll
    for (int j = 0; j < 16; ++j) dst[j] = (_Float16)x0[j];
  }
  __syncthreads();

  // ---------- GEMM1: h1 = relu(x0 @ W1 + b1), [32,512]x[512,256] ----------
  {
    v8f acc[2][2] = {};
    const int nt0 = wave * 2;             // this wave's 2 N-tiles
    for (int ks = 0; ks < 16; ++ks) {
      const int kb = ks * 32;
      __builtin_prefetch(W1t + (long)nt0 * 16 * 512 + kb + 64, 0, 3);
      v16h a0 = load_frag_f16(x0h, 0,  kb, XS, lane);
      v16h a1 = load_frag_f16(x0h, 16, kb, XS, lane);
      v16h b0 = load_frag_f16(W1t, nt0 * 16,       kb, 512, lane);
      v16h b1 = load_frag_f16(W1t, (nt0 + 1) * 16, kb, 512, lane);
      acc[0][0] = wmma_f16(a0, b0, acc[0][0]);
      acc[1][0] = wmma_f16(a1, b0, acc[1][0]);
      acc[0][1] = wmma_f16(a0, b1, acc[0][1]);
      acc[1][1] = wmma_f16(a1, b1, acc[1][1]);
    }
#pragma unroll
    for (int nt = 0; nt < 2; ++nt) {
      const int n = (nt0 + nt) * 16 + l16;   // C/D layout: lane -> N
      const float b = db1[n];
#pragma unroll
      for (int mt = 0; mt < 2; ++mt) {
#pragma unroll
        for (int v = 0; v < 8; ++v) {        // VGPR v -> M (hi half: +8)
          float val = acc[mt][nt][v] + b;
          val = val > 0.f ? val : 0.f;
          h1[(mt * 16 + hi * 8 + v) * H1S + n] = (_Float16)val;
        }
      }
    }
  }
  __syncthreads();

  // ---------- GEMM2: h2 = relu(h1 @ W2 + b2), [32,256]x[256,128] ----------
  {
    v8f acc[2] = {};
    for (int ks = 0; ks < 8; ++ks) {
      const int kb = ks * 32;
      v16h a0 = load_frag_f16(h1, 0,  kb, H1S, lane);
      v16h a1 = load_frag_f16(h1, 16, kb, H1S, lane);
      v16h b  = load_frag_f16(W2t, wave * 16, kb, 256, lane);
      acc[0] = wmma_f16(a0, b, acc[0]);
      acc[1] = wmma_f16(a1, b, acc[1]);
    }
    const int n = wave * 16 + l16;
    const float bb = db2[n];
#pragma unroll
    for (int mt = 0; mt < 2; ++mt) {
#pragma unroll
      for (int v = 0; v < 8; ++v) {
        float val = acc[mt][v] + bb;
        val = val > 0.f ? val : 0.f;
        h2[(mt * 16 + hi * 8 + v) * H2S + n] = (_Float16)val;
      }
    }
  }
  __syncthreads();

  // ---------- GEMM3 + head: h3 = relu(h2 @ W3 + b3); out += h3.Wo[512:] ---
  {
    const int mt = wave & 1;              // M tile (0..1)
    const int nt = wave >> 1;             // N tile (0..3), 64 cols total
    v8f acc = {};
    for (int ks = 0; ks < 4; ++ks) {
      const int kb = ks * 32;
      v16h a = load_frag_f16(h2, mt * 16, kb, H2S, lane);
      v16h b = load_frag_f16(W3t, nt * 16, kb, 128, lane);
      acc = wmma_f16(a, b, acc);
    }
    const int n = nt * 16 + l16;
    const float bb = db3[n];
    const float wo = Wo[512 + n];
#pragma unroll
    for (int v = 0; v < 8; ++v) {
      float val = acc[v] + bb;
      val = val > 0.f ? val : 0.f;
      float c = val * wo;
      // reduce over the 16 lanes holding this output row (same half)
#pragma unroll
      for (int m = 8; m >= 1; m >>= 1) c += __shfl_xor(c, m, 32);
      if (l16 == 0) atomicAdd(&rowAcc[mt * 16 + hi * 8 + v], c);
    }
  }
  __syncthreads();

  if (tid < ROWS) out[rowBase + tid] = rowAcc[tid] + bo[0];
}

extern "C" void kernel_launch(void* const* d_in, const int* in_sizes, int n_in,
                              void* d_out, int out_size, void* d_ws, size_t ws_size,
                              hipStream_t stream) {
  const int*   user     = (const int*)  d_in[0];
  const int*   item     = (const int*)  d_in[1];
  const float* user_emb = (const float*)d_in[2];
  const float* item_emb = (const float*)d_in[3];
  const float* cw1      = (const float*)d_in[4];
  const float* cw2      = (const float*)d_in[5];
  const float* dW1      = (const float*)d_in[6];
  const float* db1      = (const float*)d_in[7];
  const float* dW2      = (const float*)d_in[8];
  const float* db2      = (const float*)d_in[9];
  const float* dW3      = (const float*)d_in[10];
  const float* db3      = (const float*)d_in[11];
  const float* Wo       = (const float*)d_in[12];
  const float* bo       = (const float*)d_in[13];
  float* out = (float*)d_out;

  // Workspace: f16 transposed weights ([N][K]) — 256KB + 64KB + 16KB
  _Float16* W1t = (_Float16*)d_ws;
  _Float16* W2t = (_Float16*)((char*)d_ws + (size_t)512 * 256 * 2);
  _Float16* W3t = (_Float16*)((char*)d_ws + (size_t)512 * 256 * 2 + (size_t)256 * 128 * 2);

  cvt_transpose<<<(512 * 256 + 255) / 256, 256, 0, stream>>>(dW1, W1t, 512, 256);
  cvt_transpose<<<(256 * 128 + 255) / 256, 256, 0, stream>>>(dW2, W2t, 256, 128);
  cvt_transpose<<<(128 * 64  + 255) / 256, 256, 0, stream>>>(dW3, W3t, 128, 64);

  dcn_fused<<<16384 / ROWS, 256, 0, stream>>>(user, item, user_emb, item_emb,
                                              cw1, cw2, W1t, db1, W2t, db2,
                                              W3t, db3, Wo, bo, out);
}
